// TextEncoder_63256278336259
// MI455X (gfx1250) — matrix operand order repaired and checked by reading, not compiled, not run on hardware
//
#include <hip/hip_runtime.h>
#include <hip/hip_bf16.h>

typedef _Float16 v16h __attribute__((ext_vector_type(16)));
typedef _Float16 v8h  __attribute__((ext_vector_type(8)));
typedef float    v8f  __attribute__((ext_vector_type(8)));

#define VOCAB 50000
#define E     300
#define EP    320          // E padded to multiple of 32 (WMMA K)
#define H     256
#define G     768          // 3*H
#define B     64
#define S     512
#define ROWS  (B * S)      // 32768

// ---------------------------------------------------------------------------
// Kernel 1: embedding gather + mask. Writes f32 emb output, f32 mask output,
// and a K-padded f16 copy of emb for the WMMA GEMM.
// ---------------------------------------------------------------------------
__global__ __launch_bounds__(256) void embed_kernel(
    const int* __restrict__ inds, const int* __restrict__ lens,
    const float* __restrict__ table,
    float* __restrict__ out_emb, float* __restrict__ out_mask,
    _Float16* __restrict__ embH)
{
    const int tid      = threadIdx.x;
    const int lane     = tid & 31;
    const int rowInBlk = tid >> 5;                                  // 8 rows/block
    const long long row = (long long)blockIdx.x * 8 + rowInBlk;     // < ROWS
    const int b = (int)(row >> 9);          // row / S
    const int s = (int)(row & (S - 1));     // row % S
    const float m = (s < lens[b]) ? 1.0f : 0.0f;

    const long long v = inds[row];
    const float* __restrict__ src = table + v * (long long)E;
    float*    dstF = out_emb + row * (long long)E;
    _Float16* dstH = embH   + row * (long long)EP;

    for (int e = lane; e < EP; e += 32) {
        float val = (e < E) ? src[e] * m : 0.0f;
        if (e < E) dstF[e] = val;
        dstH[e] = (_Float16)val;
    }
    if (lane == 0) out_mask[row] = m;
}

// ---------------------------------------------------------------------------
// Kernel 2: convert weights to f16 (w_ih padded K 300->320).
// ---------------------------------------------------------------------------
__global__ __launch_bounds__(256) void convw_kernel(
    const float* __restrict__ w_ih, const float* __restrict__ w_hh,
    _Float16* __restrict__ w_ihH, _Float16* __restrict__ w_hhH)
{
    const int g = blockIdx.x;   // 0..G-1
    for (int e = threadIdx.x; e < EP; e += 256)
        w_ihH[(long long)g * EP + e] =
            (_Float16)((e < E) ? w_ih[(long long)g * E + e] : 0.0f);
    for (int k = threadIdx.x; k < H; k += 256)
        w_hhH[(long long)g * H + k] = (_Float16)w_hh[(long long)g * H + k];
}

// ---------------------------------------------------------------------------
// Kernel 3: gx = embH @ w_ihH^T + b_ih   ([32768 x 320] x [320 x 768], f32 acc)
// One wave computes a 16x64 output strip (1 M-tile x 4 N-tiles), K-loop of 10.
// Lane fragment layout used here (k-contiguous, same k order for A and B):
//   A (16x32 f16): lane l -> row l&15, K chunks {k0+off..+7} and {k0+8+off..+7},
//                  off = (l<16)?0:16  -> two contiguous v8h loads.
//   B (32x16 f16): lane l -> col l&15, K range k0+((l<16)?0:16) .. +15
//                  -> one contiguous v16h load (k-contiguous == row-major w).
// ---------------------------------------------------------------------------
__global__ __launch_bounds__(256) void gx_gemm_kernel(
    const _Float16* __restrict__ embH, const _Float16* __restrict__ w_ihH,
    const float* __restrict__ b_ih, float* __restrict__ gx)
{
    const int tid  = threadIdx.x;
    const int lane = tid & 31;
    const int wave = blockIdx.x * 8 + (tid >> 5);   // 0..24575
    const int mt   = wave / 12;                     // M tile 0..2047
    const int ng   = wave % 12;                     // group of 4 N tiles

    const int arow  = (lane & 15);
    const int aoff  = (lane < 16) ? 0 : 16;
    const int nlane = lane & 15;
    const int kb16  = (lane < 16) ? 0 : 16;

    const _Float16* __restrict__ aptr = embH + ((long long)mt * 16 + arow) * EP;

    v8f acc[4] = {};
    for (int k0 = 0; k0 < EP; k0 += 32) {
        v8h alo = *(const v8h*)(aptr + k0 + aoff);
        v8h ahi = *(const v8h*)(aptr + k0 + 8 + aoff);
        v16h a = __builtin_shufflevector(alo, ahi,
                    0,1,2,3,4,5,6,7,8,9,10,11,12,13,14,15);
        #pragma unroll
        for (int tI = 0; tI < 4; ++tI) {
            const int g = (ng * 4 + tI) * 16 + nlane;
            v16h bm = *(const v16h*)(w_ihH + (long long)g * EP + k0 + kb16);
            acc[tI] = __builtin_amdgcn_wmma_f32_16x16x32_f16(
                false, a, false, bm, (short)0, acc[tI], false, false);
        }
    }

    // Epilogue: D layout -> lane l element r is (row mt*16 + r + offm, col g)
    const int offm = (lane < 16) ? 0 : 8;
    #pragma unroll
    for (int tI = 0; tI < 4; ++tI) {
        const int g = (ng * 4 + tI) * 16 + nlane;
        const float bias = b_ih[g];
        #pragma unroll
        for (int r = 0; r < 8; ++r)
            gx[((long long)mt * 16 + offm + r) * G + g] = acc[tI][r] + bias;
    }
}

// ---------------------------------------------------------------------------
// Kernel 4: serial GRU. 4 workgroups x 512 threads; WG handles 16 batch rows
// (batch rows are independent -> no inter-WG sync). h lives in LDS (f32 + f16);
// per step: gh = h @ w_hh^T via WMMA (16 waves x 3 N-tiles x 8 K-steps),
// then elementwise gates r,z,n (PyTorch order) update h; feats/hT written.
// ---------------------------------------------------------------------------
__global__ __launch_bounds__(512) void gru_kernel(
    const _Float16* __restrict__ w_hhH, const float* __restrict__ b_hh,
    const float* __restrict__ gx, const int* __restrict__ lens,
    float* __restrict__ feats, float* __restrict__ hiddens)
{
    __shared__ float    hF[16][H + 8];    // f32 master state  (16.9 KB)
    __shared__ _Float16 hH[16][H + 8];    // f16 WMMA copy     ( 8.4 KB)
    __shared__ _Float16 ghS[16][G];       // gh staging        (24.6 KB)

    const int tid  = threadIdx.x;
    const int lane = tid & 31;
    const int wave = tid >> 5;            // 0..15

    // WMMA fragment lane geometry
    const int arow  = lane & 15;
    const int aoff  = (lane < 16) ? 0 : 16;
    const int nlane = lane & 15;
    const int kb16  = (lane < 16) ? 0 : 16;
    const int offm  = (lane < 16) ? 0 : 8;

    // Gate-stage assignment: wave w owns local batch row w; lane owns 8 cols.
    const int grow = wave;
    const int gj   = lane * 8;
    const int bg   = blockIdx.x * 16 + grow;     // global batch row
    const int len  = lens[bg];
    const float* __restrict__ gxrow = gx + (long long)bg * S * G;

    // zero-init state
    for (int i = tid; i < 16 * (H + 8); i += 512) {
        (&hF[0][0])[i] = 0.0f;
        (&hH[0][0])[i] = (_Float16)0.0f;
    }

    // preload per-thread bias slices (fixed j range across all timesteps)
    float bhr[8], bhz[8], bhn[8];
    #pragma unroll
    for (int u = 0; u < 8; ++u) {
        bhr[u] = b_hh[gj + u];
        bhz[u] = b_hh[H + gj + u];
        bhn[u] = b_hh[2 * H + gj + u];
    }
    __syncthreads();

    for (int t = 0; t < S; ++t) {
        // ---- gh = h @ w_hh^T : this wave computes N tiles 3*wave .. 3*wave+2
        v8f acc[3] = {};
        #pragma unroll 2
        for (int k0 = 0; k0 < H; k0 += 32) {
            v8h alo = *(const v8h*)(&hH[arow][k0 + aoff]);
            v8h ahi = *(const v8h*)(&hH[arow][k0 + 8 + aoff]);
            v16h a = __builtin_shufflevector(alo, ahi,
                        0,1,2,3,4,5,6,7,8,9,10,11,12,13,14,15);
            #pragma unroll
            for (int tI = 0; tI < 3; ++tI) {
                const int g = (wave * 3 + tI) * 16 + nlane;
                v16h bm = *(const v16h*)(w_hhH + (long long)g * H + k0 + kb16);
                acc[tI] = __builtin_amdgcn_wmma_f32_16x16x32_f16(
                    false, a, false, bm, (short)0, acc[tI], false, false);
            }
        }
        #pragma unroll
        for (int tI = 0; tI < 3; ++tI) {
            const int g = (wave * 3 + tI) * 16 + nlane;
            #pragma unroll
            for (int r = 0; r < 8; ++r)
                ghS[offm + r][g] = (_Float16)acc[tI][r];
        }
        __syncthreads();

        // ---- gates + state update (each thread: row grow, cols gj..gj+7)
        const float* __restrict__ gxt = gxrow + (long long)t * G;
        if (t + 1 < S) __builtin_prefetch(gxt + G, 0, 3);   // global_prefetch_b8
        const bool valid = (t < len);
        float* __restrict__ fout = feats + ((long long)bg * S + t) * H + gj;

        #pragma unroll
        for (int u = 0; u < 8; ++u) {
            const int j = gj + u;
            const float ghr = (float)ghS[grow][j]         + bhr[u];
            const float ghz = (float)ghS[grow][H + j]     + bhz[u];
            const float ghn = (float)ghS[grow][2 * H + j] + bhn[u];
            const float rg = 1.0f / (1.0f + __expf(-(gxt[j] + ghr)));
            const float zg = 1.0f / (1.0f + __expf(-(gxt[H + j] + ghz)));
            const float ng = tanhf(gxt[2 * H + j] + rg * ghn);
            const float hprev = hF[grow][j];
            const float hnew  = (1.0f - zg) * ng + zg * hprev;
            const float hout  = valid ? hnew : hprev;
            hF[grow][j] = hout;
            hH[grow][j] = (_Float16)hout;
            fout[u] = valid ? hnew : 0.0f;
        }
        __syncthreads();
    }

    // final hidden state hT -> hiddens[1, B, H]
    #pragma unroll
    for (int u = 0; u < 8; ++u)
        hiddens[(long long)bg * H + gj + u] = hF[grow][gj + u];
}

// ---------------------------------------------------------------------------
extern "C" void kernel_launch(void* const* d_in, const int* in_sizes, int n_in,
                              void* d_out, int out_size, void* d_ws, size_t ws_size,
                              hipStream_t stream)
{
    (void)in_sizes; (void)n_in; (void)out_size; (void)ws_size;

    const int*   inds  = (const int*)d_in[0];
    const int*   lens  = (const int*)d_in[1];
    const float* table = (const float*)d_in[2];
    const float* w_ih  = (const float*)d_in[3];
    const float* w_hh  = (const float*)d_in[4];
    const float* b_ih  = (const float*)d_in[5];
    const float* b_hh  = (const float*)d_in[6];

    float* out       = (float*)d_out;
    float* out_emb   = out;                                  // [B,S,E]
    float* out_feats = out_emb + (size_t)B * S * E;          // [B,S,H]
    float* out_mask  = out_feats + (size_t)B * S * H;        // [B,S]
    float* out_hid   = out_mask + (size_t)B * S;             // [1,B,H]

    char* ws = (char*)d_ws;
    size_t o = 0;
    _Float16* embH = (_Float16*)(ws + o); o += (size_t)ROWS * EP * sizeof(_Float16);
    _Float16* wihH = (_Float16*)(ws + o); o += (size_t)G * EP * sizeof(_Float16);
    _Float16* whhH = (_Float16*)(ws + o); o += (size_t)G * H  * sizeof(_Float16);
    o = (o + 255) & ~(size_t)255;
    float* gx = (float*)(ws + o);        // [ROWS, G] f32 (~96 MB)

    embed_kernel  <<<ROWS / 8, 256, 0, stream>>>(inds, lens, table,
                                                 out_emb, out_mask, embH);
    convw_kernel  <<<G,        256, 0, stream>>>(w_ih, w_hh, wihH, whhH);
    gx_gemm_kernel<<<3072,     256, 0, stream>>>(embH, wihH, b_ih, gx);
    gru_kernel    <<<4,        512, 0, stream>>>(whhH, b_hh, gx, lens,
                                                 out_feats, out_hid);
}
